// graphLayer_46213848105194
// MI455X (gfx1250) — compile-verified, matrix operand order checked by reading, not run on hardware
//
#include <hip/hip_runtime.h>
#include <hip/hip_bf16.h>

// ---------------------------------------------------------------------------
// Problem constants (match the reference)
// ---------------------------------------------------------------------------
constexpr int BB = 8;       // batch
constexpr int NT = 1024;    // tokens
constexpr int DM = 512;     // model dim
constexpr int HH = 8;       // heads
constexpr int HD = 64;      // head dim
constexpr float NEGV = -1e30f;
constexpr float RSCALE = 0.044194173824159216f;  // 1/sqrt(512)

typedef __attribute__((ext_vector_type(16))) _Float16     v16h;
typedef __attribute__((ext_vector_type(8)))  _Float16     v8h;
typedef __attribute__((ext_vector_type(8)))  float        v8f;
typedef __attribute__((ext_vector_type(4)))  unsigned int v4u;
typedef __attribute__((ext_vector_type(8)))  int          v8i;
typedef __attribute__((ext_vector_type(4)))  int          v4i;

// ---------------------------------------------------------------------------
// WMMA helper: D = A(16x32 f16) * B(32x16 f16) + C(f32)
// ---------------------------------------------------------------------------
__device__ __forceinline__ v8f wmma_f16(v16h a, v16h b, v8f c) {
  return __builtin_amdgcn_wmma_f32_16x16x32_f16(
      /*neg_a=*/false, a, /*neg_b=*/false, b,
      /*c_mod=*/(short)0, c, /*reuse_a=*/false, /*reuse_b=*/false);
}

// f16 operand fetch: ISA 16-bit A/B layout wants, per lane, 8 contiguous
// halves at kLo (elements 0..7) and 8 at kHi (elements 8..15).
// Works for global and LDS pointers (generic addressing).
__device__ __forceinline__ v16h ld16(const _Float16* p, int kLo, int kHi) {
  const v8h lo = *reinterpret_cast<const v8h*>(p + kLo);
  const v8h hi = *reinterpret_cast<const v8h*>(p + kHi);
  return __builtin_shufflevector(lo, hi, 0, 1, 2, 3, 4, 5, 6, 7,
                                 8, 9, 10, 11, 12, 13, 14, 15);
}

// fp32 source variant (for LDS-resident softmax probabilities)
__device__ __forceinline__ v16h pack16(const float* p, int kLo, int kHi) {
  const float4 x0 = *reinterpret_cast<const float4*>(p + kLo);
  const float4 x1 = *reinterpret_cast<const float4*>(p + kLo + 4);
  const float4 x2 = *reinterpret_cast<const float4*>(p + kHi);
  const float4 x3 = *reinterpret_cast<const float4*>(p + kHi + 4);
  v16h v;
  v[0]  = (_Float16)x0.x; v[1]  = (_Float16)x0.y;
  v[2]  = (_Float16)x0.z; v[3]  = (_Float16)x0.w;
  v[4]  = (_Float16)x1.x; v[5]  = (_Float16)x1.y;
  v[6]  = (_Float16)x1.z; v[7]  = (_Float16)x1.w;
  v[8]  = (_Float16)x2.x; v[9]  = (_Float16)x2.y;
  v[10] = (_Float16)x2.z; v[11] = (_Float16)x2.w;
  v[12] = (_Float16)x3.x; v[13] = (_Float16)x3.y;
  v[14] = (_Float16)x3.z; v[15] = (_Float16)x3.w;
  return v;
}

// ---------------------------------------------------------------------------
// One-time fp32 -> f16 conversion (inputs + weights)
// ---------------------------------------------------------------------------
__global__ __launch_bounds__(256) void cvt_f32_to_f16(
    const float* __restrict__ s, _Float16* __restrict__ d, int n) {
  const int i = blockIdx.x * 256 + threadIdx.x;
  if (i < n) d[i] = (_Float16)s[i];
}

// ---------------------------------------------------------------------------
// f16 GEMM: C16 = f16( A[M,K] @ W[N,K]^T + bias[N] )   (torch Linear form)
// Block = 128 threads (4 waves), 64x64 output tile. The shared weight tile
// W[n0:n0+64, 0:K] (f16, <=64KB) is staged into LDS by ONE Tensor Data Mover
// DMA per block (descriptor per CDNA5 ISA section 8.3/8.4), then the K-loop
// runs from ds_load_b128. A-rows are wave-exclusive and stream from L2.
// headSplit=1 re-tiles the output into [B,H,N,HD] for the attention kernel.
// ---------------------------------------------------------------------------
__global__ __launch_bounds__(128) void gemm_f16_wmma(
    const _Float16* __restrict__ A, const _Float16* __restrict__ W,
    const float* __restrict__ bias, _Float16* __restrict__ C16,
    int M, int N, int K, int headSplit) {
  __shared__ _Float16 wtile[64 * 512];  // sole LDS object -> group offset 0

  const int lane  = threadIdx.x & 31;
  const int wave  = threadIdx.x >> 5;
  const int laneM = lane & 15;   // M (A) / N (B) index of this lane
  const int laneH = lane >> 4;   // K half select
  const int m0 = blockIdx.x * 64 + wave * 16;
  const int n0 = blockIdx.y * 64;

  // ---- TDM stage: W[n0:n0+64, 0:K] -> LDS (wave 0 issues; EXEC ignored) ----
  if (wave == 0) {
    const unsigned long long ga =
        (unsigned long long)(const void*)(W + (size_t)n0 * K);
    v4u g0;
    g0[0] = 1u;                                              // count=1
    g0[1] = 0u;                                              // lds_addr = 0
    g0[2] = (unsigned)ga;                                    // global_addr lo
    g0[3] = (unsigned)((ga >> 32) & 0x01FFFFFFu) | 0x80000000u;  // hi | type=2
    v8i g1;
    g1[0] = 0x10000;                                         // data_size=2B
    g1[1] = (K & 0xFFFF) << 16;                              // tensor_dim0 lo
    g1[2] = ((K >> 16) & 0xFFFF) | ((N & 0xFFFF) << 16);     // d0 hi | d1 lo
    g1[3] = ((N >> 16) & 0xFFFF) | ((K & 0xFFFF) << 16);     // d1 hi | tile_d0=K
    g1[4] = 64;                                              // tile_d1=64, d2=0
    g1[5] = K;                                               // dim0_stride lo
    g1[6] = 0;
    g1[7] = 0;
    const v4i gz4 = {0, 0, 0, 0};
    const v8i gz8 = {0, 0, 0, 0, 0, 0, 0, 0};
    __builtin_amdgcn_tensor_load_to_lds(g0, g1, gz4, gz4, gz8, 0);
    __builtin_amdgcn_s_wait_tensorcnt(0);
  }
  // The TDM builtin's LDS write is not modeled against `wtile`; without this,
  // the compiler folds the (apparently never-written) LDS reads to poison and
  // deletes them. Escape the address + memory clobber = zero-cost fix.
  {
    unsigned long long wp = (unsigned long long)(void*)wtile;
    asm volatile("" : "+v"(wp) : : "memory");
  }
  __syncthreads();

  const _Float16* aRow = A + (size_t)(m0 + laneM) * K;
  const _Float16* w0 = wtile + (size_t)(laneM +  0) * K;
  const _Float16* w1 = wtile + (size_t)(laneM + 16) * K;
  const _Float16* w2 = wtile + (size_t)(laneM + 32) * K;
  const _Float16* w3 = wtile + (size_t)(laneM + 48) * K;

  v8f acc[4] = {{}, {}, {}, {}};
  for (int k0 = 0; k0 < K; k0 += 32) {
    const int kLo = k0 + laneH * 8;
    const int kHi = k0 + 16 + laneH * 8;
    const v16h a = ld16(aRow, kLo, kHi);
    acc[0] = wmma_f16(a, ld16(w0, kLo, kHi), acc[0]);
    acc[1] = wmma_f16(a, ld16(w1, kLo, kHi), acc[1]);
    acc[2] = wmma_f16(a, ld16(w2, kLo, kHi), acc[2]);
    acc[3] = wmma_f16(a, ld16(w3, kLo, kHi), acc[3]);
  }

  // C/D layout: VGPR r -> row = laneH*8 + r, col = laneM (within 16x16 tile)
#pragma unroll
  for (int nt = 0; nt < 4; ++nt) {
    const int n = n0 + nt * 16 + laneM;
    const float bv = bias[n];
#pragma unroll
    for (int r = 0; r < 8; ++r) {
      const int m = m0 + laneH * 8 + r;
      const float v = acc[nt][r] + bv;
      size_t idx;
      if (headSplit) {
        const int bi = m >> 10, row = m & (NT - 1);
        const int hi = n >> 6,  dd  = n & (HD - 1);
        idx = (((size_t)bi * HH + hi) * NT + row) * HD + dd;
      } else {
        idx = (size_t)m * N + n;
      }
      C16[idx] = (_Float16)v;
    }
  }
}

// ---------------------------------------------------------------------------
// Masked multi-head attention, f16-native operands. One wave per
// (b, h, 16-row tile); grid = (N/16, H, B). Score stripe [16x1024] fp32 in
// 64KB LDS. head/tail/info are [B,H,N,HD] f16; output attn is [B,N,D] f16.
// ---------------------------------------------------------------------------
__global__ __launch_bounds__(32) void attn_kernel(
    const _Float16* __restrict__ head, const _Float16* __restrict__ tail,
    const _Float16* __restrict__ info, const int* __restrict__ graph,
    _Float16* __restrict__ out) {
  __shared__ float sc[16 * 1024];  // 64 KB

  const int lane  = threadIdx.x;
  const int laneM = lane & 15;
  const int laneH = lane >> 4;
  const int b = blockIdx.z;
  const int h = blockIdx.y;
  const int row0 = blockIdx.x * 16;

  const _Float16* headB = head + ((size_t)b * HH + h) * NT * HD;
  const _Float16* tailB = tail + ((size_t)b * HH + h) * NT * HD;
  const _Float16* infoB = info + ((size_t)b * HH + h) * NT * HD;

  // ---- Stage 1: scores = head @ tail^T / sqrt(D), masked -> LDS ----
  const _Float16* aRow = headB + (size_t)(row0 + laneM) * HD;
  const v16h a0 = ld16(aRow, laneH * 8,      16 + laneH * 8);
  const v16h a1 = ld16(aRow, 32 + laneH * 8, 48 + laneH * 8);

  for (int j = 0; j < NT / 16; ++j) {
    const int coln = j * 16 + laneM;
    const _Float16* bRow = tailB + (size_t)coln * HD;
    const v16h b0 = ld16(bRow, laneH * 8,      16 + laneH * 8);
    const v16h b1 = ld16(bRow, 32 + laneH * 8, 48 + laneH * 8);
    v8f c = {};
    c = wmma_f16(a0, b0, c);
    c = wmma_f16(a1, b1, c);
#pragma unroll
    for (int r = 0; r < 8; ++r) {
      const int rowm = laneH * 8 + r;
      const int g = graph[((size_t)b * NT + row0 + rowm) * NT + coln];
      sc[rowm * NT + coln] = (g == 0) ? NEGV : c[r] * RSCALE;
    }
  }
  __syncthreads();

  // ---- Stage 2: row softmax (exp(NEG - max) underflows to exactly 0,
  //               reproducing the reference's re-masking to 0) ----
  for (int r = 0; r < 16; ++r) {
    float* rowp = sc + r * NT;
    float mx = -3.0e38f;
    for (int c = lane; c < NT; c += 32) mx = fmaxf(mx, rowp[c]);
#pragma unroll
    for (int off = 16; off; off >>= 1) mx = fmaxf(mx, __shfl_xor(mx, off, 32));
    float s = 0.0f;
    for (int c = lane; c < NT; c += 32) {
      const float e = __expf(rowp[c] - mx);
      rowp[c] = e;
      s += e;
    }
#pragma unroll
    for (int off = 16; off; off >>= 1) s += __shfl_xor(s, off, 32);
    const float inv = 1.0f / s;
    for (int c = lane; c < NT; c += 32) rowp[c] *= inv;
  }
  __syncthreads();

  // ---- Stage 3: attn = P[16,1024] @ info[1024,64] ----
  v8f acc[4] = {{}, {}, {}, {}};
  const float* pRow = sc + (size_t)laneM * NT;
  for (int kt = 0; kt < NT / 32; ++kt) {
    const int k0 = kt * 32;
    const v16h a = pack16(pRow, k0 + laneH * 8, k0 + 16 + laneH * 8);
#pragma unroll
    for (int dt = 0; dt < 4; ++dt) {
      const int d = dt * 16 + laneM;
      v16h bb;
#pragma unroll
      for (int i = 0; i < 8; ++i) {  // k-strided, lane-contiguous loads
        bb[i]     = infoB[(size_t)(k0 + laneH * 8 + i) * HD + d];
        bb[i + 8] = infoB[(size_t)(k0 + 16 + laneH * 8 + i) * HD + d];
      }
      acc[dt] = wmma_f16(a, bb, acc[dt]);
    }
  }

  // write back f16 in [B,N,D] layout (heads re-interleaved)
  _Float16* outB = out + ((size_t)b * NT + row0) * DM + (size_t)h * HD;
#pragma unroll
  for (int dt = 0; dt < 4; ++dt) {
    const int d = dt * 16 + laneM;
#pragma unroll
    for (int r = 0; r < 8; ++r) {
      outB[(size_t)(laneH * 8 + r) * DM + d] = (_Float16)acc[dt][r];
    }
  }
}

// ---------------------------------------------------------------------------
// GRU gates + LayerNorm (f16 gates, fp32 math). One block per token row.
// ---------------------------------------------------------------------------
__global__ __launch_bounds__(128) void gru_ln_kernel(
    const _Float16* __restrict__ gi, const _Float16* __restrict__ gh,
    const _Float16* __restrict__ hprev, const float* __restrict__ ln_g,
    const float* __restrict__ ln_b, float* __restrict__ out) {
  const int row  = blockIdx.x;
  const int tid  = threadIdx.x;
  const int lane = tid & 31;
  const int wid  = tid >> 5;

  const _Float16* giR = gi + (size_t)row * (3 * DM);
  const _Float16* ghR = gh + (size_t)row * (3 * DM);
  const _Float16* hR  = hprev + (size_t)row * DM;

  float hn[4];
  float sum = 0.0f;
#pragma unroll
  for (int e = 0; e < 4; ++e) {
    const int idx = e * 128 + tid;
    const float r =
        1.0f / (1.0f + __expf(-((float)giR[idx] + (float)ghR[idx])));
    const float z = 1.0f /
        (1.0f + __expf(-((float)giR[DM + idx] + (float)ghR[DM + idx])));
    const float n =
        tanhf((float)giR[2 * DM + idx] + r * (float)ghR[2 * DM + idx]);
    const float v = (1.0f - z) * n + z * (float)hR[idx];
    hn[e] = v;
    sum += v;
  }

  __shared__ float red[4];
#pragma unroll
  for (int off = 16; off; off >>= 1) sum += __shfl_xor(sum, off, 32);
  if (lane == 0) red[wid] = sum;
  __syncthreads();
  const float mu = (red[0] + red[1] + red[2] + red[3]) * (1.0f / DM);
  __syncthreads();

  float vs = 0.0f;
#pragma unroll
  for (int e = 0; e < 4; ++e) {
    const float dlt = hn[e] - mu;
    vs += dlt * dlt;
  }
#pragma unroll
  for (int off = 16; off; off >>= 1) vs += __shfl_xor(vs, off, 32);
  if (lane == 0) red[wid] = vs;
  __syncthreads();
  const float inv =
      rsqrtf((red[0] + red[1] + red[2] + red[3]) * (1.0f / DM) + 1e-5f);

#pragma unroll
  for (int e = 0; e < 4; ++e) {
    const int idx = e * 128 + tid;
    out[(size_t)row * DM + idx] = (hn[e] - mu) * inv * ln_g[idx] + ln_b[idx];
  }
}

// ---------------------------------------------------------------------------
// Host-side orchestration
// ---------------------------------------------------------------------------
extern "C" void kernel_launch(void* const* d_in, const int* in_sizes, int n_in,
                              void* d_out, int out_size, void* d_ws, size_t ws_size,
                              hipStream_t stream) {
  const float* inputs   = (const float*)d_in[0];
  const float* memory   = (const float*)d_in[1];
  const int*   graph    = (const int*)d_in[2];
  const float* w_head_w = (const float*)d_in[3];
  const float* w_head_b = (const float*)d_in[4];
  const float* w_tail_w = (const float*)d_in[5];
  const float* w_tail_b = (const float*)d_in[6];
  const float* w_info_w = (const float*)d_in[7];
  const float* w_info_b = (const float*)d_in[8];
  const float* w_comb_w = (const float*)d_in[9];
  const float* w_comb_b = (const float*)d_in[10];
  const float* gru_w_ih = (const float*)d_in[11];
  const float* gru_w_hh = (const float*)d_in[12];
  const float* gru_b_ih = (const float*)d_in[13];
  const float* gru_b_hh = (const float*)d_in[14];
  const float* ln_g     = (const float*)d_in[15];
  const float* ln_b     = (const float*)d_in[16];
  float* outp = (float*)d_out;

  const size_t MT = (size_t)BB * NT;  // 8192 token rows
  _Float16* p = (_Float16*)d_ws;      // peak ~109 MB of scratch (f16-native)
  _Float16* in16   = p; p += MT * DM;
  _Float16* mem16  = p; p += MT * DM;
  _Float16* wh16   = p; p += (size_t)DM * DM;
  _Float16* wt16   = p; p += (size_t)DM * DM;
  _Float16* wi16   = p; p += (size_t)DM * DM;
  _Float16* wc16   = p; p += (size_t)DM * DM;
  _Float16* wih16  = p; p += (size_t)3 * DM * DM;
  _Float16* whh16  = p; p += (size_t)3 * DM * DM;
  _Float16* head16 = p; p += MT * DM;
  _Float16* tail16 = p; p += MT * DM;
  _Float16* info16 = p; p += MT * DM;
  _Float16* attn16 = p; p += MT * DM;
  _Float16* hcb16  = p; p += MT * DM;
  _Float16* gi16   = p; p += MT * 3 * DM;
  _Float16* gh16   = p; p += MT * 3 * DM;

  // ---- one-time fp32 -> f16 conversions ----
  auto cvt = [&](const float* s, _Float16* d, size_t n) {
    cvt_f32_to_f16<<<(int)((n + 255) / 256), 256, 0, stream>>>(s, d, (int)n);
  };
  cvt(inputs,   in16,  MT * DM);
  cvt(memory,   mem16, MT * DM);
  cvt(w_head_w, wh16,  (size_t)DM * DM);
  cvt(w_tail_w, wt16,  (size_t)DM * DM);
  cvt(w_info_w, wi16,  (size_t)DM * DM);
  cvt(w_comb_w, wc16,  (size_t)DM * DM);
  cvt(gru_w_ih, wih16, (size_t)3 * DM * DM);
  cvt(gru_w_hh, whh16, (size_t)3 * DM * DM);

  const dim3 blk(128);
  const dim3 g512((int)(MT / 64), DM / 64);        // (128, 8)
  const dim3 g1536((int)(MT / 64), (3 * DM) / 64); // (128, 24)

  // projections -> head-split f16 layout for attention
  gemm_f16_wmma<<<g512, blk, 0, stream>>>(in16, wh16, w_head_b, head16,
                                          (int)MT, DM, DM, 1);
  gemm_f16_wmma<<<g512, blk, 0, stream>>>(mem16, wt16, w_tail_b, tail16,
                                          (int)MT, DM, DM, 1);
  gemm_f16_wmma<<<g512, blk, 0, stream>>>(mem16, wi16, w_info_b, info16,
                                          (int)MT, DM, DM, 1);

  // masked multi-head attention
  const dim3 ga(NT / 16, HH, BB);
  attn_kernel<<<ga, 32, 0, stream>>>(head16, tail16, info16, graph, attn16);

  // combine linear -> initial hidden state h (f16)
  gemm_f16_wmma<<<g512, blk, 0, stream>>>(attn16, wc16, w_comb_b, hcb16,
                                          (int)MT, DM, DM, 0);

  // GRU gate GEMMs
  gemm_f16_wmma<<<g1536, blk, 0, stream>>>(in16, wih16, gru_b_ih, gi16,
                                           (int)MT, 3 * DM, DM, 0);
  gemm_f16_wmma<<<g1536, blk, 0, stream>>>(hcb16, whh16, gru_b_hh, gh16,
                                           (int)MT, 3 * DM, DM, 0);

  // GRU elementwise + LayerNorm
  gru_ln_kernel<<<(int)MT, blk, 0, stream>>>(gi16, gh16, hcb16, ln_g, ln_b,
                                             outp);
}
